// EdgeBlock_83631603188044
// MI455X (gfx1250) — matrix operand order repaired and checked by reading, not compiled, not run on hardware
//
#include <hip/hip_runtime.h>
#include <hip/hip_bf16.h>

typedef __attribute__((ext_vector_type(16))) __bf16 v16bf;
typedef __attribute__((ext_vector_type(8)))  __bf16 bf16x8;
typedef __attribute__((ext_vector_type(8)))  float  v8f;

#define LDH 72  // bf16 row stride for hidden tiles   (144B, 16B aligned)
#define LDW 72  // bf16 row stride for W^T in LDS     (144B, 16B aligned)

// A-matrix fragment (16x32 MxK bf16) from row-major W^T [M][LDW] in LDS.
// Layout (contiguous k order, same as B here): lane L -> M = L&15, h = L>>4,
//   element e -> K = h*16 + e.
// q = &WT[mt*16 + (L&15)][kstep*32 + h*16] : two 16B chunks at q and q+8.
__device__ __forceinline__ v16bf afrag_wT(const __bf16* q) {
  bf16x8 lo = *(const bf16x8*)q;
  bf16x8 hi = *(const bf16x8*)(q + 8);
  return __builtin_shufflevector(lo, hi, 0, 1, 2, 3, 4, 5, 6, 7,
                                 8, 9, 10, 11, 12, 13, 14, 15);
}

// B-matrix fragment (32x16 KxN bf16) of H^T from a row-major H tile in LDS.
// Layout: lane L -> N(edge) = L&15, element e -> K = (L>>4)*16 + e.
// p = &H[L&15][kstep*32 + (L>>4)*16] : 16 contiguous bf16 -> two ds_load_b128.
__device__ __forceinline__ v16bf bfrag_lds(const __bf16* p) {
  bf16x8 lo = *(const bf16x8*)p;
  bf16x8 hi = *(const bf16x8*)(p + 8);
  return __builtin_shufflevector(lo, hi, 0, 1, 2, 3, 4, 5, 6, 7,
                                 8, 9, 10, 11, 12, 13, 14, 15);
}

// D-fragment store: lane L holds edge N = L&15, VGPR j -> feature h*8+j.
// relu(acc) packed to 8 contiguous bf16 -> one ds_store_b128.
// (Bias already applied via the WMMA C operand.)
__device__ __forceinline__ void store_relu8(__bf16* p, v8f acc) {
  bf16x8 v;
#pragma unroll
  for (int j = 0; j < 8; ++j) {
    float r = acc[j];
    r = r > 0.0f ? r : 0.0f;  // single v_max_num
    v[j] = (__bf16)r;
  }
  *(bf16x8*)p = v;
}

#define WMMA_BF16(A, B, C) \
  __builtin_amdgcn_wmma_f32_16x16x32_bf16(false, (A), false, (B), (short)0, (C), false, false)

#define WAVE_FENCE() __builtin_amdgcn_fence(__ATOMIC_ACQ_REL, "wavefront")

__global__ __launch_bounds__(256) void edgeblock_wmma_kernel(
    const float* __restrict__ edges, const float* __restrict__ nodes,
    const float* __restrict__ globals_, const float* __restrict__ W0,
    const float* __restrict__ b0, const float* __restrict__ W1,
    const float* __restrict__ b1, const float* __restrict__ W2,
    const float* __restrict__ b2, const int* __restrict__ senders,
    const int* __restrict__ receivers, const int* __restrict__ gids,
    float* __restrict__ out, int nE) {
  __shared__ __attribute__((aligned(16))) __bf16 sW0T[64 * LDW];
  __shared__ __attribute__((aligned(16))) __bf16 sW1T[64 * LDW];
  __shared__ __attribute__((aligned(16))) __bf16 sW2T[16 * LDW];
  __shared__ __attribute__((aligned(16))) float sB0[64];
  __shared__ __attribute__((aligned(16))) float sB1[64];
  __shared__ __attribute__((aligned(16))) float sB2[16];
  __shared__ __attribute__((aligned(16))) __bf16 sH[8][2][16 * LDH];

  const int tid  = threadIdx.x;
  const int lane = tid & 31;
  const int wid  = tid >> 5;
  const int m    = lane & 15;  // edge-in-tile (N) / A-row select
  const int h    = lane >> 4;  // lane-half

  // Cooperative: stage W^T (bf16) in LDS. Global reads coalesced; LDS writes
  // stride 144B across adjacent threads (bank-spread).
  for (int i = tid; i < 64 * 64; i += 256) {
    const int k = i >> 6, n = i & 63;
    sW0T[n * LDW + k] = (__bf16)W0[i];
    sW1T[n * LDW + k] = (__bf16)W1[i];
  }
  for (int i = tid; i < 64 * 16; i += 256) {
    const int k = i >> 4, n = i & 15;
    sW2T[n * LDW + k] = (__bf16)W2[i];
  }
  if (tid < 64) { sB0[tid] = b0[tid]; sB1[tid] = b1[tid]; }
  if (tid < 16) sB2[tid] = b2[tid];
  __syncthreads();

  // Register-resident A fragments of W0^T / W1^T / W2^T (vectorized b128 loads).
  v16bf fa0[4][2], fa1[4][2], fa2[2];
#pragma unroll
  for (int mt = 0; mt < 4; ++mt) {
#pragma unroll
    for (int ks = 0; ks < 2; ++ks) {
      const int off = (mt * 16 + m) * LDW + ks * 32 + h * 16;
      fa0[mt][ks] = afrag_wT(sW0T + off);
      fa1[mt][ks] = afrag_wT(sW1T + off);
    }
  }
#pragma unroll
  for (int ks = 0; ks < 2; ++ks)
    fa2[ks] = afrag_wT(sW2T + m * LDW + ks * 32 + h * 16);

  // Register-resident bias accumulator-init vectors: D row = mt*16 + h*8 + j.
  v8f bv0[4], bv1[4], bv2;
#pragma unroll
  for (int mt = 0; mt < 4; ++mt) {
    const float4* p0 = (const float4*)(sB0 + mt * 16 + h * 8);
    const float4* p1 = (const float4*)(sB1 + mt * 16 + h * 8);
    float4 a = p0[0], b = p0[1], c = p1[0], d = p1[1];
    bv0[mt][0] = a.x; bv0[mt][1] = a.y; bv0[mt][2] = a.z; bv0[mt][3] = a.w;
    bv0[mt][4] = b.x; bv0[mt][5] = b.y; bv0[mt][6] = b.z; bv0[mt][7] = b.w;
    bv1[mt][0] = c.x; bv1[mt][1] = c.y; bv1[mt][2] = c.z; bv1[mt][3] = c.w;
    bv1[mt][4] = d.x; bv1[mt][5] = d.y; bv1[mt][6] = d.z; bv1[mt][7] = d.w;
  }
  {
    const float4* p2 = (const float4*)(sB2 + h * 8);
    float4 a = p2[0], b = p2[1];
    bv2[0] = a.x; bv2[1] = a.y; bv2[2] = a.z; bv2[3] = a.w;
    bv2[4] = b.x; bv2[5] = b.y; bv2[6] = b.z; bv2[7] = b.w;
  }

  const int nTiles = (nE + 15) >> 4;
  const int stride = gridDim.x * 8;

  __bf16* H0 = sH[wid][0];
  __bf16* H1 = sH[wid][1];

  for (int tile = blockIdx.x * 8 + wid; tile < nTiles; tile += stride) {
    const int base = tile * 16;
    int eidx = base + m;
    if (eidx >= nE) eidx = nE - 1;

    // Prefetch next tile's edge rows (1KB contiguous) into cache.
    {
      const int nt2 = tile + stride;
      if (nt2 < nTiles) {
        const float* pf = edges + (size_t)nt2 * 256 + lane * 8;
        __builtin_prefetch(pf, 0, 0);
      }
    }

    // X^T B-fragments: each lane loads one full 16-float source row.
    //   bx0: lanes 0-15 -> edges[eidx], lanes 16-31 -> nodes[receivers[eidx]]
    //   bx1: lanes 0-15 -> nodes[senders[eidx]], lanes 16-31 -> globals_[gid]
    int idxA = eidx;
    if (h) idxA = receivers[eidx];
    const int* pB = h ? gids : senders;
    const int idxB = pB[eidx];
    const float4* row0 =
        (const float4*)(h ? nodes + (size_t)idxA * 16 : edges + (size_t)idxA * 16);
    const float4* row1 =
        (const float4*)(h ? globals_ + (size_t)idxB * 16 : nodes + (size_t)idxB * 16);
    float4 x0 = row0[0], x1 = row0[1], x2 = row0[2], x3 = row0[3];
    float4 y0 = row1[0], y1 = row1[1], y2 = row1[2], y3 = row1[3];

    v16bf bx0, bx1;
    bx0[0]  = (__bf16)x0.x; bx0[1]  = (__bf16)x0.y; bx0[2]  = (__bf16)x0.z; bx0[3]  = (__bf16)x0.w;
    bx0[4]  = (__bf16)x1.x; bx0[5]  = (__bf16)x1.y; bx0[6]  = (__bf16)x1.z; bx0[7]  = (__bf16)x1.w;
    bx0[8]  = (__bf16)x2.x; bx0[9]  = (__bf16)x2.y; bx0[10] = (__bf16)x2.z; bx0[11] = (__bf16)x2.w;
    bx0[12] = (__bf16)x3.x; bx0[13] = (__bf16)x3.y; bx0[14] = (__bf16)x3.z; bx0[15] = (__bf16)x3.w;
    bx1[0]  = (__bf16)y0.x; bx1[1]  = (__bf16)y0.y; bx1[2]  = (__bf16)y0.z; bx1[3]  = (__bf16)y0.w;
    bx1[4]  = (__bf16)y1.x; bx1[5]  = (__bf16)y1.y; bx1[6]  = (__bf16)y1.z; bx1[7]  = (__bf16)y1.w;
    bx1[8]  = (__bf16)y2.x; bx1[9]  = (__bf16)y2.y; bx1[10] = (__bf16)y2.z; bx1[11] = (__bf16)y2.w;
    bx1[12] = (__bf16)y3.x; bx1[13] = (__bf16)y3.y; bx1[14] = (__bf16)y3.z; bx1[15] = (__bf16)y3.w;

    // ---- Layer 0: H0^T = relu(W0^T @ X^T + b0) ; bias rides in C ----
#pragma unroll
    for (int mt = 0; mt < 4; ++mt) {
      v8f acc = bv0[mt];
      acc = WMMA_BF16(fa0[mt][0], bx0, acc);
      acc = WMMA_BF16(fa0[mt][1], bx1, acc);
      store_relu8(H0 + m * LDH + mt * 16 + h * 8, acc);
    }
    WAVE_FENCE();

    // ---- Layer 1: H1^T = relu(W1^T @ H0^T + b1) ----
    {
      const __bf16* hp = H0 + m * LDH + h * 16;
      v16bf c0 = bfrag_lds(hp);
      v16bf c1 = bfrag_lds(hp + 32);
      WAVE_FENCE();
#pragma unroll
      for (int mt = 0; mt < 4; ++mt) {
        v8f acc = bv1[mt];
        acc = WMMA_BF16(fa1[mt][0], c0, acc);
        acc = WMMA_BF16(fa1[mt][1], c1, acc);
        store_relu8(H1 + m * LDH + mt * 16 + h * 8, acc);
      }
    }
    WAVE_FENCE();

    // ---- Layer 2: Out^T = W2^T @ H1^T + b2 ; coalesced float4 stores ----
    {
      const __bf16* hp = H1 + m * LDH + h * 16;
      v16bf c0 = bfrag_lds(hp);
      v16bf c1 = bfrag_lds(hp + 32);
      v8f acc = bv2;
      acc = WMMA_BF16(fa2[0], c0, acc);
      acc = WMMA_BF16(fa2[1], c1, acc);
      const int orow = base + m;  // this lane's edge row
      if (orow < nE) {
        float4 o0, o1;
        o0.x = acc[0]; o0.y = acc[1]; o0.z = acc[2]; o0.w = acc[3];
        o1.x = acc[4]; o1.y = acc[5]; o1.z = acc[6]; o1.w = acc[7];
        float4* op = (float4*)(out + (size_t)orow * 16 + h * 8);
        op[0] = o0;
        op[1] = o1;
      }
    }
    WAVE_FENCE();
  }
}

extern "C" void kernel_launch(void* const* d_in, const int* in_sizes, int n_in,
                              void* d_out, int out_size, void* d_ws, size_t ws_size,
                              hipStream_t stream) {
  const float* edges    = (const float*)d_in[0];
  const float* nodes    = (const float*)d_in[1];
  const float* globals_ = (const float*)d_in[2];
  const float* W0       = (const float*)d_in[3];
  const float* b0       = (const float*)d_in[4];
  const float* W1       = (const float*)d_in[5];
  const float* b1       = (const float*)d_in[6];
  const float* W2       = (const float*)d_in[7];
  const float* b2       = (const float*)d_in[8];
  const int* senders    = (const int*)d_in[9];
  const int* receivers  = (const int*)d_in[10];
  const int* gids       = (const int*)d_in[11];
  float* out            = (float*)d_out;

  const int nE     = in_sizes[0] / 16;  // D_EDGE == 16
  const int nTiles = (nE + 15) / 16;
  int blocks = 1024;  // 8 waves/block, grid-stride over 16-edge tiles
  const int maxBlocks = (nTiles + 7) / 8;
  if (blocks > maxBlocks) blocks = maxBlocks;
  if (blocks < 1) blocks = 1;

  edgeblock_wmma_kernel<<<blocks, 256, 0, stream>>>(
      edges, nodes, globals_, W0, b0, W1, b1, W2, b2,
      senders, receivers, gids, out, nE);
}
